// Informer_Encoder_8718783610930
// MI455X (gfx1250) — compile-verified, matrix operand order checked by reading, not run on hardware
//
#include <hip/hip_runtime.h>
#include <hip/hip_bf16.h>

typedef __attribute__((ext_vector_type(16))) _Float16 v16h;
typedef __attribute__((ext_vector_type(8)))  _Float16 v8h;
typedef __attribute__((ext_vector_type(8)))  float    v8f;
typedef unsigned int u32x4 __attribute__((ext_vector_type(4)));
typedef int          i32x4 __attribute__((ext_vector_type(4)));
typedef int          i32x8 __attribute__((ext_vector_type(8)));

#define D_MODEL 128
#define N_HEADS 8
#define EH      16
#define E_LAYERS 5
#define D_FF    64
#define REP     128
#define BATCH   8
#define LSEQ    4096
#define MTOT    (BATCH * LSEQ)          // 32768 rows
#define CSEL    27                      // min(3*ceil(ln 4096), 4096)

#if __has_builtin(__builtin_amdgcn_tensor_load_to_lds) && \
    __has_builtin(__builtin_amdgcn_s_wait_tensorcnt)
#define USE_TDM 1
#else
#define USE_TDM 0
#endif

// ---------------------------------------------------------------------------
// utility
// ---------------------------------------------------------------------------
__device__ __forceinline__ unsigned hmix(unsigned x) {
    x ^= x >> 16; x *= 0x7feb352du;
    x ^= x >> 15; x *= 0x846ca68bu;
    x ^= x >> 16; return x;
}

__device__ __forceinline__ float wave_sum(float v) {
    #pragma unroll
    for (int off = 16; off; off >>= 1) v += __shfl_xor(v, off, 32);
    return v;
}

// ---------------------------------------------------------------------------
// f32 -> f16 bulk convert (weights, once per launch)
// ---------------------------------------------------------------------------
__global__ void cvt_f32_f16(const float* __restrict__ src, _Float16* __restrict__ dst, int n) {
    int i = blockIdx.x * 256 + threadIdx.x;
    if (i < n) dst[i] = (_Float16)src[i];
}

// ---------------------------------------------------------------------------
// input layout change: x[b][d][l] -> h[(b*L+l)*128 + d]  (f32 + f16 mirror)
// ---------------------------------------------------------------------------
__global__ void transpose_in(const float* __restrict__ x, float* __restrict__ h,
                             _Float16* __restrict__ h16) {
    int i = blockIdx.x * blockDim.x + threadIdx.x;     // over B*D*L = 4M
    int l = i & (LSEQ - 1);
    int d = (i >> 12) & (D_MODEL - 1);
    int b = i >> 19;
    float v = x[i];
    size_t o = ((size_t)(b * LSEQ + l)) * D_MODEL + d;
    h[o] = v;
    h16[o] = (_Float16)v;
}

// ---------------------------------------------------------------------------
// WMMA GEMM:  Y[M x N] = X[M x K] * W^T (+bias)
//   X, W are f16.  W stored [N x K] row-major, or COLB=true -> W is [K x N].
//   MT = M-subtiles per block. The A stripe (MT*16 x K) is DMA'd into LDS
//   once per block via the Tensor Data Mover; waves read fragments from LDS.
//   GELU: exact gelu. LN: fused residual + layernorm (N must be 128).
//   TRANS: scatter-store as out[b][n][l]. W32/W16: which outputs to write.
// grid.x = M/(16*MT), blockDim = (N/16)*32
// ---------------------------------------------------------------------------
template <int N, int K, int MT, bool GELU, bool LN, bool COLB, bool TRANS,
          bool W32, bool W16>
__global__ void gemm_wmma(const _Float16* __restrict__ X,
                          const _Float16* __restrict__ Wm,
                          const float* __restrict__ bias,
                          const float* __restrict__ R,     // residual (LN path)
                          const float* __restrict__ lnw,
                          const float* __restrict__ lnb,
                          float* __restrict__ Y,
                          _Float16* __restrict__ Y16) {
    const int wave = threadIdx.x >> 5;
    const int lane = threadIdx.x & 31;
    const int r16  = lane & 15;
    const int hs   = lane >> 4;          // half-wave select
    const int mBase = blockIdx.x * (MT * 16);
    const int nBase = wave << 4;
    const int k0off = hs ? 8 : 0;

    // prefetch next block's A stripe into L2 (gfx1250 global_prefetch_b8)
    __builtin_prefetch(X + (size_t)(mBase + MT * 16 + r16) * K, 0, 1);

#if USE_TDM
    // ---- TDM: DMA the (MT*16 x K) f16 A stripe into LDS, one op per block --
    __shared__ alignas(16) _Float16 atile[MT * 16 * K];
    if (threadIdx.x < 32) {
        unsigned lds_off = (unsigned)(uintptr_t)(void*)atile;
        unsigned long long ga =
            (unsigned long long)(uintptr_t)(X + (size_t)mBase * K);
        u32x4 g0;
        g0[0] = 1u;                                        // count=1, user D#
        g0[1] = lds_off;                                   // lds_addr
        g0[2] = (unsigned)(ga & 0xFFFFFFFFu);              // global_addr[31:0]
        g0[3] = (unsigned)((ga >> 32) & 0x01FFFFFFu) |     // global_addr[56:32]
                (2u << 30);                                // type=2 (image)
        i32x8 g1;
        g1[0] = 0x00010000;                    // wg_mask=0, data_size=1 (2B)
        g1[1] = (int)((unsigned)K << 16);      // tensor_dim0[15:0] (K elems)
        g1[2] = (int)((unsigned)(MT * 16) << 16); // dim0 hi=0 | tensor_dim1 lo
        g1[3] = (int)((unsigned)K << 16);      // dim1 hi=0 | tile_dim0 = K
        g1[4] = MT * 16;                       // tile_dim1 = rows, tile_dim2=0
        g1[5] = K;                             // tensor_dim0_stride = row pitch
        g1[6] = 0;
        g1[7] = 0;
        i32x4 z4 = {0, 0, 0, 0};
#if __clang_major__ >= 23
        i32x8 z8 = {0, 0, 0, 0, 0, 0, 0, 0};
        __builtin_amdgcn_tensor_load_to_lds(g0, g1, z4, z4, z8, 0);
#else
        __builtin_amdgcn_tensor_load_to_lds(g0, g1, z4, z4, 0);
#endif
        __builtin_amdgcn_s_wait_tensorcnt(0);
    }
    __syncthreads();
#endif

    v8f acc[MT];
    #pragma unroll
    for (int mt = 0; mt < MT; ++mt) acc[mt] = (v8f){};

    #pragma unroll
    for (int kb = 0; kb < K; kb += 32) {
        // B fragment: column n = nBase + r16; loaded once, reused for MT tiles
        v16h bf;
        if constexpr (!COLB) {
            const _Float16* wrow = Wm + (size_t)(nBase + r16) * K + kb + k0off;
            v8h lo = *(const v8h*)(wrow);
            v8h hi = *(const v8h*)(wrow + 16);
            #pragma unroll
            for (int i = 0; i < 8; ++i) { bf[i] = lo[i]; bf[8 + i] = hi[i]; }
        } else {
            const _Float16* wcol = Wm + (nBase + r16);
            #pragma unroll
            for (int i = 0; i < 8; ++i) {
                bf[i]     = wcol[(size_t)(kb + k0off + i) * N];
                bf[8 + i] = wcol[(size_t)(kb + k0off + 16 + i) * N];
            }
        }
        #pragma unroll
        for (int mt = 0; mt < MT; ++mt) {
#if USE_TDM
            const _Float16* xr = atile + (mt * 16 + r16) * K + kb + k0off;
#else
            const _Float16* xr = X + (size_t)(mBase + mt * 16 + r16) * K + kb + k0off;
#endif
            v8h lo = *(const v8h*)(xr);
            v8h hi = *(const v8h*)(xr + 16);
            v16h a;
            #pragma unroll
            for (int i = 0; i < 8; ++i) { a[i] = lo[i]; a[8 + i] = hi[i]; }
            acc[mt] = __builtin_amdgcn_wmma_f32_16x16x32_f16(false, a, false, bf,
                                                             (short)0, acc[mt],
                                                             false, false);
        }
    }

    const int n = nBase + r16;
    const float bb = bias[n];
    #pragma unroll
    for (int mt = 0; mt < MT; ++mt) {
        #pragma unroll
        for (int r = 0; r < 8; ++r) {
            float v = acc[mt][r] + bb;
            if constexpr (GELU) v = 0.5f * v * (1.0f + erff(v * 0.70710678118654752f));
            acc[mt][r] = v;
        }
    }

    if constexpr (LN) {
        // stage full (MT*16) x 128 tile (post residual), then cooperative LN
        __shared__ float tile[MT * 16][D_MODEL + 4];
        #pragma unroll
        for (int mt = 0; mt < MT; ++mt) {
            #pragma unroll
            for (int r = 0; r < 8; ++r) {
                int row = mt * 16 + hs * 8 + r;
                size_t m = (size_t)(mBase + row);
                tile[row][n] = acc[mt][r] + R[m * D_MODEL + n];
            }
        }
        __syncthreads();
        // 256 threads = 16 groups of 16, one group per row, MT passes
        int g = threadIdx.x >> 4;
        int j = threadIdx.x & 15;
        #pragma unroll
        for (int rr = 0; rr < MT; ++rr) {
            int row = rr * 16 + g;
            float vals[8], s = 0.f, ss = 0.f;
            #pragma unroll
            for (int i = 0; i < 8; ++i) {
                float v = tile[row][j * 8 + i];
                vals[i] = v; s += v; ss += v * v;
            }
            #pragma unroll
            for (int off = 8; off; off >>= 1) {
                s  += __shfl_xor(s,  off, 16);
                ss += __shfl_xor(ss, off, 16);
            }
            float mean = s * (1.0f / D_MODEL);
            float var  = ss * (1.0f / D_MODEL) - mean * mean;
            float rstd = rsqrtf(var + 1e-5f);
            size_t m = (size_t)(mBase + row);
            #pragma unroll
            for (int i = 0; i < 8; ++i) {
                int col = j * 8 + i;
                float o = (vals[i] - mean) * rstd * lnw[col] + lnb[col];
                if constexpr (W32) Y[m * D_MODEL + col] = o;
                if constexpr (W16) Y16[m * D_MODEL + col] = (_Float16)o;
            }
        }
    } else {
        #pragma unroll
        for (int mt = 0; mt < MT; ++mt) {
            #pragma unroll
            for (int r = 0; r < 8; ++r) {
                size_t m = (size_t)(mBase + mt * 16 + hs * 8 + r);
                if constexpr (TRANS) {
                    int b = (int)(m >> 12);
                    int l = (int)(m & (LSEQ - 1));
                    Y[((size_t)(b * REP + n) << 12) + l] = acc[mt][r];
                } else {
                    if constexpr (W32) Y[m * N + n] = acc[mt][r];
                    if constexpr (W16) Y16[m * N + n] = (_Float16)acc[mt][r];
                }
            }
        }
    }
}

// ---------------------------------------------------------------------------
// ProbSparse attention, one block per (batch, head); 256 threads
// Q,K,V are f32 [M x 128]; C is f16 [M x 128]; head hh = cols 16*hh..16*hh+15
// ---------------------------------------------------------------------------
__global__ void prob_attn(const float* __restrict__ Q, const float* __restrict__ Kb,
                          const float* __restrict__ V, _Float16* __restrict__ C,
                          unsigned seed) {
    const int b  = blockIdx.x >> 3;
    const int hh = blockIdx.x & 7;
    const size_t rowBase = (size_t)b * LSEQ;
    const int co = hh * EH;
    const int t  = threadIdx.x;
    const int lane = t & 31;

    __shared__ float Ms[LSEQ];          // 16KB: sparsity measure, then scores
    __shared__ float vmean[EH];
    __shared__ float redv[256];
    __shared__ int   redi[256];
    __shared__ int   top[CSEL];
    __shared__ float supd[EH];
    __shared__ float ssum;

    // ---- V mean over sequence ----
    float lm[EH];
    #pragma unroll
    for (int e = 0; e < EH; ++e) lm[e] = 0.f;
    for (int l = t; l < LSEQ; l += 256) {
        const float* vr = V + (rowBase + l) * D_MODEL + co;
        #pragma unroll
        for (int e = 0; e < EH; ++e) lm[e] += vr[e];
    }
    if (t < EH) vmean[t] = 0.f;
    __syncthreads();
    #pragma unroll
    for (int e = 0; e < EH; ++e) {
        float s = wave_sum(lm[e]);
        if (lane == 0) atomicAdd(&vmean[e], s);
    }
    __syncthreads();
    if (t < EH) vmean[t] *= (1.0f / LSEQ);
    __syncthreads();

    // ---- sparsity measure M over sampled keys ----
    for (int l = t; l < LSEQ; l += 256) {
        const float* qr = Q + (rowBase + l) * D_MODEL + co;
        float qv[EH];
        #pragma unroll
        for (int e = 0; e < EH; ++e) qv[e] = qr[e];
        float mx = -3.0e38f, sm = 0.f;
        for (int u = 0; u < CSEL; ++u) {
            int j = hmix(seed ^ (unsigned)(l * CSEL + u)) & (LSEQ - 1);
            const float* kr = Kb + (rowBase + j) * D_MODEL + co;
            float d = 0.f;
            #pragma unroll
            for (int e = 0; e < EH; ++e) d += qv[e] * kr[e];
            mx = fmaxf(mx, d); sm += d;
        }
        Ms[l] = mx - sm * (1.0f / LSEQ);
    }
    __syncthreads();

    // ---- iterative top-CSEL ----
    for (int i = 0; i < CSEL; ++i) {
        float bm = -3.0e38f; int bi = 0;
        for (int l = t; l < LSEQ; l += 256) {
            float v = Ms[l];
            if (v > bm) { bm = v; bi = l; }
        }
        redv[t] = bm; redi[t] = bi;
        __syncthreads();
        for (int s = 128; s; s >>= 1) {
            if (t < s && redv[t + s] > redv[t]) { redv[t] = redv[t + s]; redi[t] = redi[t + s]; }
            __syncthreads();
        }
        if (t == 0) { top[i] = redi[0]; Ms[redi[0]] = -3.0e38f; }
        __syncthreads();
    }

    // ---- context = mean(V) everywhere (f16 store) ----
    for (int l = t; l < LSEQ; l += 256) {
        _Float16* cr = C + (rowBase + l) * D_MODEL + co;
        #pragma unroll
        for (int e = 0; e < EH; ++e) cr[e] = (_Float16)vmean[e];
    }
    __syncthreads();

    // ---- full softmax attention for the selected queries ----
    for (int u = 0; u < CSEL; ++u) {
        int lq = top[u];
        const float* qr = Q + (rowBase + lq) * D_MODEL + co;
        float qv[EH];
        #pragma unroll
        for (int e = 0; e < EH; ++e) qv[e] = qr[e] * 0.25f;   // 1/sqrt(16)
        float mx = -3.0e38f;
        for (int j = t; j < LSEQ; j += 256) {
            const float* kr = Kb + (rowBase + j) * D_MODEL + co;
            float d = 0.f;
            #pragma unroll
            for (int e = 0; e < EH; ++e) d += qv[e] * kr[e];
            Ms[j] = d; mx = fmaxf(mx, d);
        }
        redv[t] = mx;
        __syncthreads();
        for (int s = 128; s; s >>= 1) {
            if (t < s) redv[t] = fmaxf(redv[t], redv[t + s]);
            __syncthreads();
        }
        float gmax = redv[0];
        if (t < EH) supd[t] = 0.f;
        if (t == 0) ssum = 0.f;
        __syncthreads();
        float ls = 0.f, lupd[EH];
        #pragma unroll
        for (int e = 0; e < EH; ++e) lupd[e] = 0.f;
        for (int j = t; j < LSEQ; j += 256) {
            float p = __expf(Ms[j] - gmax);
            ls += p;
            const float* vr = V + (rowBase + j) * D_MODEL + co;
            #pragma unroll
            for (int e = 0; e < EH; ++e) lupd[e] += p * vr[e];
        }
        float s = wave_sum(ls);
        if (lane == 0) atomicAdd(&ssum, s);
        #pragma unroll
        for (int e = 0; e < EH; ++e) {
            float se = wave_sum(lupd[e]);
            if (lane == 0) atomicAdd(&supd[e], se);
        }
        __syncthreads();
        if (t < EH) {
            _Float16* cr = C + (rowBase + lq) * D_MODEL + co;
            cr[t] = (_Float16)(supd[t] / ssum);
        }
        __syncthreads();
    }
}

// ---------------------------------------------------------------------------
// final layernorm: f32 in, f16 out (feeds last WMMA GEMM)
// ---------------------------------------------------------------------------
__global__ void ln_kernel(const float* __restrict__ h, _Float16* __restrict__ o16,
                          const float* __restrict__ w, const float* __restrict__ b) {
    int g = threadIdx.x >> 4;
    int j = threadIdx.x & 15;
    size_t m = (size_t)blockIdx.x * 16 + g;
    const float* row = h + m * D_MODEL;
    float vals[8], s = 0.f, ss = 0.f;
    #pragma unroll
    for (int i = 0; i < 8; ++i) {
        float v = row[j * 8 + i];
        vals[i] = v; s += v; ss += v * v;
    }
    #pragma unroll
    for (int off = 8; off; off >>= 1) {
        s  += __shfl_xor(s,  off, 16);
        ss += __shfl_xor(ss, off, 16);
    }
    float mean = s * (1.0f / D_MODEL);
    float var  = ss * (1.0f / D_MODEL) - mean * mean;
    float rstd = rsqrtf(var + 1e-5f);
    #pragma unroll
    for (int i = 0; i < 8; ++i) {
        int col = j * 8 + i;
        o16[m * D_MODEL + col] = (_Float16)((vals[i] - mean) * rstd * w[col] + b[col]);
    }
}

// ---------------------------------------------------------------------------
extern "C" void kernel_launch(void* const* d_in, const int* in_sizes, int n_in,
                              void* d_out, int out_size, void* d_ws, size_t ws_size,
                              hipStream_t stream) {
    const float* x    = (const float*)d_in[0];
    const float* Wq   = (const float*)d_in[1];
    const float* bq   = (const float*)d_in[2];
    const float* Wk   = (const float*)d_in[3];
    const float* bk   = (const float*)d_in[4];
    const float* Wv   = (const float*)d_in[5];
    const float* bv   = (const float*)d_in[6];
    const float* Wo   = (const float*)d_in[7];
    const float* bo   = (const float*)d_in[8];
    const float* W1   = (const float*)d_in[9];
    const float* b1   = (const float*)d_in[10];
    const float* W2   = (const float*)d_in[11];
    const float* b2   = (const float*)d_in[12];
    const float* ln1w = (const float*)d_in[13];
    const float* ln1b = (const float*)d_in[14];
    const float* ln2w = (const float*)d_in[15];
    const float* ln2b = (const float*)d_in[16];
    const float* lnfw = (const float*)d_in[17];
    const float* lnfb = (const float*)d_in[18];
    const float* Wp   = (const float*)d_in[19];
    const float* bp   = (const float*)d_in[20];

    const size_t S = (size_t)MTOT * D_MODEL;     // 4M elements
    float* ws = (float*)d_ws;
    float* h = ws;
    float* q = ws + 1 * S;
    float* k = ws + 2 * S;
    float* v = ws + 3 * S;

    _Float16* h16   = (_Float16*)(ws + 4 * S);
    _Float16* ctx16 = h16 + S;
    _Float16* ff16  = ctx16 + S;                 // uses S/2
    _Float16* wq16  = ff16 + S / 2;
    _Float16* wk16  = wq16 + E_LAYERS * D_MODEL * D_MODEL;
    _Float16* wv16  = wk16 + E_LAYERS * D_MODEL * D_MODEL;
    _Float16* wo16  = wv16 + E_LAYERS * D_MODEL * D_MODEL;
    _Float16* w116  = wo16 + E_LAYERS * D_MODEL * D_MODEL;
    _Float16* w216  = w116 + E_LAYERS * D_FF * D_MODEL;
    _Float16* wp16  = w216 + E_LAYERS * D_MODEL * D_FF;

    // one-time weight conversion to f16
    const int NW4 = E_LAYERS * D_MODEL * D_MODEL;   // 81920
    const int NWF = E_LAYERS * D_FF * D_MODEL;      // 40960
    cvt_f32_f16<<<(NW4 + 255) / 256, 256, 0, stream>>>(Wq, wq16, NW4);
    cvt_f32_f16<<<(NW4 + 255) / 256, 256, 0, stream>>>(Wk, wk16, NW4);
    cvt_f32_f16<<<(NW4 + 255) / 256, 256, 0, stream>>>(Wv, wv16, NW4);
    cvt_f32_f16<<<(NW4 + 255) / 256, 256, 0, stream>>>(Wo, wo16, NW4);
    cvt_f32_f16<<<(NWF + 255) / 256, 256, 0, stream>>>(W1, w116, NWF);
    cvt_f32_f16<<<(NWF + 255) / 256, 256, 0, stream>>>(W2, w216, NWF);
    cvt_f32_f16<<<(D_MODEL * REP + 255) / 256, 256, 0, stream>>>(Wp, wp16, D_MODEL * REP);

    transpose_in<<<(BATCH * D_MODEL * LSEQ) / 256, 256, 0, stream>>>(x, h, h16);

    const int GBLK = MTOT / 64;                  // 512 blocks (MT=4)

    for (int l = 0; l < E_LAYERS; ++l) {
        const _Float16* wq = wq16 + (size_t)l * D_MODEL * D_MODEL;
        const _Float16* wk = wk16 + (size_t)l * D_MODEL * D_MODEL;
        const _Float16* wv = wv16 + (size_t)l * D_MODEL * D_MODEL;
        const _Float16* wo = wo16 + (size_t)l * D_MODEL * D_MODEL;
        const _Float16* w1 = w116 + (size_t)l * D_FF * D_MODEL;
        const _Float16* w2 = w216 + (size_t)l * D_MODEL * D_FF;

        // Q/K/V projections (f32 out for attention)
        gemm_wmma<128, 128, 4, false, false, false, false, true, false>
            <<<GBLK, 256, 0, stream>>>(h16, wq, bq + l * D_MODEL,
                                       nullptr, nullptr, nullptr, q, nullptr);
        gemm_wmma<128, 128, 4, false, false, false, false, true, false>
            <<<GBLK, 256, 0, stream>>>(h16, wk, bk + l * D_MODEL,
                                       nullptr, nullptr, nullptr, k, nullptr);
        gemm_wmma<128, 128, 4, false, false, false, false, true, false>
            <<<GBLK, 256, 0, stream>>>(h16, wv, bv + l * D_MODEL,
                                       nullptr, nullptr, nullptr, v, nullptr);

        // ProbSparse attention (writes f16 context)
        prob_attn<<<BATCH * N_HEADS, 256, 0, stream>>>(q, k, v, ctx16,
                                                       0x9e3779b9u * (unsigned)(l + 1));

        // output projection + residual + LN1 (writes h f32 + h16 f16)
        gemm_wmma<128, 128, 4, false, true, false, false, true, true>
            <<<GBLK, 256, 0, stream>>>(ctx16, wo, bo + l * D_MODEL, h,
                                       ln1w + l * D_MODEL, ln1b + l * D_MODEL, h, h16);

        // FFN: gelu(h @ W1^T + b1), f16-only output
        gemm_wmma<64, 128, 4, true, false, false, false, false, true>
            <<<GBLK, 128, 0, stream>>>(h16, w1, b1 + l * D_FF,
                                       nullptr, nullptr, nullptr, nullptr, ff16);

        // FFN out + residual + LN2 (writes h f32 + h16 f16)
        gemm_wmma<128, 64, 4, false, true, false, false, true, true>
            <<<GBLK, 256, 0, stream>>>(ff16, w2, b2 + l * D_MODEL, h,
                                       ln2w + l * D_MODEL, ln2b + l * D_MODEL, h, h16);
    }

    // final layernorm (f32 -> f16) + projection with transposed [B, REP, L] store
    ln_kernel<<<MTOT / 16, 256, 0, stream>>>(h, h16, lnfw, lnfb);
    gemm_wmma<128, 128, 4, false, false, true, true, true, false>
        <<<GBLK, 256, 0, stream>>>(h16, wp16, bp,
                                   nullptr, nullptr, nullptr, (float*)d_out, nullptr);
}
